// SceneGraphAttention_21629455302889
// MI455X (gfx1250) — compile-verified, matrix operand order checked by reading, not run on hardware
//
#include <hip/hip_runtime.h>

#define B_ 8
#define N_ 512
#define D_ 256
#define E_ 32
#define H_ 8
#define DH 32   // D_/H_

typedef __attribute__((ext_vector_type(16))) _Float16     v16h;
typedef __attribute__((ext_vector_type(8)))  float        v8f;
typedef __attribute__((ext_vector_type(4)))  unsigned int v4u;
typedef __attribute__((ext_vector_type(8)))  int          v8i;
typedef __attribute__((ext_vector_type(4)))  int          v4i;

#if defined(__AMDGCN__) && __has_builtin(__builtin_amdgcn_tensor_load_to_lds)
#define HAVE_TDM 1
#else
#define HAVE_TDM 0
#endif

// K-index for element pair (2t,2t+1) of a 16-bit WMMA A/B fragment (wave32).
// ISA 7.12.2: VGPR t in 0..3 -> K = hi8 + 2t..2t+1 ; t in 4..7 -> +16.
__device__ __forceinline__ int kmap(int t, int hi8) {
  return ((t >> 2) << 4) + hi8 + ((t & 3) << 1);
}

// C = A(Mx256) @ W(256x256) (+ bias). blockDim = 256 (8 waves).
// Block stages its shared 16-row A tile in LDS once (all waves reuse it);
// each wave computes one 16x16 tile.
__global__ __launch_bounds__(256)
void gemm_nt_wmma(const float* __restrict__ A, const float* __restrict__ W,
                  const float* __restrict__ bias, float* __restrict__ C) {
  __shared__ _Float16 as[16 * D_];  // 8 KB
  const int tid  = threadIdx.x;
  const int lane = tid & 31;
  const int wave = tid >> 5;
  const int hi8  = (lane >> 4) << 3;
  const int l15  = lane & 15;
  const int m0   = blockIdx.x * 16;
  const int n0   = (blockIdx.y * 8 + wave) * 16;
  const int ncol = n0 + l15;

  const float* arow = A + (size_t)m0 * D_;
  for (int idx = tid; idx < 16 * D_; idx += 256) as[idx] = (_Float16)arow[idx];
  __syncthreads();

  v8f acc = {};
  for (int k0 = 0; k0 < D_; k0 += 32) {
    v16h a, b;
#pragma unroll
    for (int t = 0; t < 8; ++t) {
      const int k = kmap(t, hi8);
      a[2 * t]     = as[l15 * D_ + k0 + k];
      a[2 * t + 1] = as[l15 * D_ + k0 + k + 1];
      const float* bp = W + (size_t)(k0 + k) * D_ + ncol;
      b[2 * t]     = (_Float16)bp[0];
      b[2 * t + 1] = (_Float16)bp[D_];
    }
    acc = __builtin_amdgcn_wmma_f32_16x16x32_f16(false, a, false, b,
                                                 (short)0, acc, false, false);
  }
  const float badd = bias ? bias[ncol] : 0.0f;
#pragma unroll
  for (int r = 0; r < 8; ++r) {
    const int row = m0 + r + hi8;
    C[(size_t)row * D_ + ncol] = acc[r] + badd;
  }
}

// Fused flash attention with edge bias + adjacency mask.
// grid = (B, N/16); block = 256 threads = 8 waves; wave w <-> head w.
// Edges tile (no dtype conversion needed) is staged by the Tensor Data Mover.
__global__ __launch_bounds__(256)
void attn_fused(const float* __restrict__ qb, const float* __restrict__ kb,
                const float* __restrict__ vb, const float* __restrict__ edges,
                const int* __restrict__ adj, const float* __restrict__ We,
                float* __restrict__ ob) {
  extern __shared__ char smem[];
  float*    es    = (float*)smem;                 // [16 i][16 j][32 e]   32 KB (at LDS base)
  float*    biass = es + 16 * 16 * 32;            // [8 h][16 i][32 j]   16 KB
  int*      adjt  = (int*)(biass + 8 * 16 * 32);  // [16 i][32 j]         2 KB
  _Float16* qs    = (_Float16*)(adjt + 16 * 32);  // [16][256]            8 KB
  _Float16* ks    = qs + 16 * D_;                 // [32][256]           16 KB
  _Float16* vs    = ks + 32 * D_;                 // [32][256]           16 KB
  _Float16* ps    = vs + 32 * D_;                 // [8 w][16 i][32 j]    8 KB

  const int b    = blockIdx.x;
  const int i0   = blockIdx.y * 16;
  const int tid  = threadIdx.x;
  const int lane = tid & 31;
  const int w    = tid >> 5;       // head index
  const int hi8  = (lane >> 4) << 3;
  const int l15  = lane & 15;
  const float scale = 0.17677669529663687f;  // (D/H)^-0.5 = 1/sqrt(32)

  // stage 16 query rows as f16
  const float* qrow = qb + ((size_t)b * N_ + i0) * D_;
  for (int idx = tid; idx < 16 * D_; idx += 256) qs[idx] = (_Float16)qrow[idx];
  __syncthreads();

  // Q A-fragment for this head (16 rows x K=32 head dims)
  v16h qf;
#pragma unroll
  for (int t = 0; t < 8; ++t) {
    const int k = kmap(t, hi8);
    qf[2 * t]     = qs[l15 * D_ + w * DH + k];
    qf[2 * t + 1] = qs[l15 * D_ + w * DH + k + 1];
  }
  // We B-fragment (32 x 16, cols >= H zero-padded)
  v16h wef;
#pragma unroll
  for (int t = 0; t < 8; ++t) {
    const int k = kmap(t, hi8);
    wef[2 * t]     = (l15 < H_) ? (_Float16)We[k * H_ + l15] : (_Float16)0.f;
    wef[2 * t + 1] = (l15 < H_) ? (_Float16)We[(k + 1) * H_ + l15] : (_Float16)0.f;
  }

  float mrowv[8], lrow[8];
  v8f o0 = {}, o1 = {};
#pragma unroll
  for (int r = 0; r < 8; ++r) { mrowv[r] = -3.0e38f; lrow[r] = 0.f; }

  for (int jp = 0; jp < N_ / 32; ++jp) {
    const int j0 = jp * 32;
    __syncthreads();  // previous iteration's LDS consumers done
    // stage K,V rows j0..j0+31 (f16) and adjacency tile
    {
      const float* kr = kb + ((size_t)b * N_ + j0) * D_;
      const float* vr = vb + ((size_t)b * N_ + j0) * D_;
      for (int idx = tid; idx < 32 * D_; idx += 256) {
        ks[idx] = (_Float16)kr[idx];
        vs[idx] = (_Float16)vr[idx];
      }
      const int* ar = adj + ((size_t)b * N_ + i0) * N_ + j0;
      for (int idx = tid; idx < 16 * 32; idx += 256)
        adjt[idx] = ar[(size_t)(idx >> 5) * N_ + (idx & 31)];
    }

    v8f sraw[2];
#pragma unroll
    for (int sub = 0; sub < 2; ++sub) {
      __syncthreads();  // ks/vs visible; es free for reuse
      // ---- stage edges[b, i0..+15, j0+sub*16..+15, 0..31] into es ----
      const float* er = edges + (((size_t)b * N_ + i0) * N_ + (j0 + sub * 16)) * E_;
#if HAVE_TDM
      if (w == 0) {  // one TDM issue per block; TDM ignores EXEC
        // D# group 0: count=1 | lds_addr | global_addr | type=2
        const unsigned long long ga = (unsigned long long)(uintptr_t)er;
        unsigned ldsa = 0;
#if __has_builtin(__builtin_amdgcn_groupstaticsize)
        ldsa = __builtin_amdgcn_groupstaticsize();  // dynamic-LDS base == es
#endif
        v4u g0;
        g0[0] = 1u;                                    // count=1, user mode
        g0[1] = ldsa;                                  // lds_addr (bytes)
        g0[2] = (unsigned)(ga & 0xFFFFFFFFull);        // global_addr[31:0]
        g0[3] = (unsigned)((ga >> 32) & 0x1FFFFFFull)  // global_addr[56:32]
                | (2u << 30);                          // type=2 ("image")
        // D# group 1: 2D tile, data_size=4B, tile 512x16, row stride N*E
        v8i g1;
        g1[0] = (int)(2u << 16);            // workgroup_mask=0, data_size=2 (4B)
        g1[1] = (int)(16384u << 16);        // tensor_dim0[15:0]=N*E=16384 (bits 63:48)
        g1[2] = (int)(512u << 16);          // tensor_dim0 hi=0, tensor_dim1[15:0]=512
        g1[3] = (int)(512u << 16);          // tensor_dim1 hi=0, tile_dim0=512
        g1[4] = 16;                         // tile_dim1=16, tile_dim2=0
        g1[5] = 16384;                      // tensor_dim0_stride[31:0] = N*E
        g1[6] = 0;                          // stride hi, tensor_dim1_stride lo (unused, 2D)
        g1[7] = 0;
        v4i z4 = {0, 0, 0, 0};
#if defined(__clang_major__) && (__clang_major__ >= 23)
        v8i z8 = {0, 0, 0, 0, 0, 0, 0, 0};
        __builtin_amdgcn_tensor_load_to_lds(g0, g1, z4, z4, z8, 0);
#else
        __builtin_amdgcn_tensor_load_to_lds(g0, g1, z4, z4, 0);
#endif
#if __has_builtin(__builtin_amdgcn_s_wait_tensorcnt)
        __builtin_amdgcn_s_wait_tensorcnt((short)0);
#else
        asm volatile("s_wait_tensorcnt 0x0" ::: "memory");
#endif
      }
#else
      for (int idx = tid; idx < 16 * 16 * 8; idx += 256) {
        const int ii = idx >> 7, rem = idx & 127, jj = rem >> 3, ve = rem & 7;
        ((float4*)es)[idx] =
            ((const float4*)(er + ((size_t)ii * N_ + jj) * E_))[ve];
      }
#endif
      __syncthreads();  // es ready (TDM waited by wave 0 before barrier)

      // edge-bias WMMA: per i, D[j_local][head] = edges[i,j,:] @ We
#pragma unroll
      for (int ii = 0; ii < 2; ++ii) {
        const int i = w + ii * 8;
        v16h ef;
#pragma unroll
        for (int t = 0; t < 8; ++t) {
          const int k = kmap(t, hi8);
          const float* ep = es + (i * 16 + l15) * 32 + k;
          ef[2 * t]     = (_Float16)ep[0];
          ef[2 * t + 1] = (_Float16)ep[1];
        }
        v8f z = {};
        v8f d = __builtin_amdgcn_wmma_f32_16x16x32_f16(false, ef, false, wef,
                                                       (short)0, z, false, false);
        if (l15 < H_) {
#pragma unroll
          for (int r = 0; r < 8; ++r)
            biass[(l15 * 16 + i) * 32 + sub * 16 + (r + hi8)] = d[r];
        }
      }
      // score WMMA: S = Q(16x32) @ K^T(32x16) for this head
      v16h kf;
#pragma unroll
      for (int t = 0; t < 8; ++t) {
        const int k = kmap(t, hi8);
        const _Float16* kp = ks + (sub * 16 + l15) * D_ + w * DH + k;
        kf[2 * t]     = kp[0];
        kf[2 * t + 1] = kp[1];
      }
      v8f z2 = {};
      sraw[sub] = __builtin_amdgcn_wmma_f32_16x16x32_f16(false, qf, false, kf,
                                                         (short)0, z2, false, false);
    }
    __syncthreads();  // bias writes visible across waves

    // masked scores + online softmax over these 32 columns
#pragma unroll
    for (int r = 0; r < 8; ++r) {
      const int iloc = r + hi8;
      float v0 = sraw[0][r] * scale + biass[(w * 16 + iloc) * 32 + l15];
      float v1 = sraw[1][r] * scale + biass[(w * 16 + iloc) * 32 + 16 + l15];
      if (adjt[iloc * 32 + l15] == 0)      v0 = -1.0e9f;
      if (adjt[iloc * 32 + 16 + l15] == 0) v1 = -1.0e9f;
      float tmax = fmaxf(v0, v1);
      tmax = fmaxf(tmax, __shfl_xor(tmax, 1));
      tmax = fmaxf(tmax, __shfl_xor(tmax, 2));
      tmax = fmaxf(tmax, __shfl_xor(tmax, 4));
      tmax = fmaxf(tmax, __shfl_xor(tmax, 8));  // stays within 16-lane half
      const float mnew = fmaxf(mrowv[r], tmax);
      const float corr = __expf(mrowv[r] - mnew);
      mrowv[r] = mnew;
      const float e0 = __expf(v0 - mnew);
      const float e1 = __expf(v1 - mnew);
      float s = e0 + e1;
      s += __shfl_xor(s, 1);
      s += __shfl_xor(s, 2);
      s += __shfl_xor(s, 4);
      s += __shfl_xor(s, 8);
      lrow[r] = lrow[r] * corr + s;
      o0[r] *= corr;
      o1[r] *= corr;
      ps[(w * 16 + iloc) * 32 + l15]      = (_Float16)e0;
      ps[(w * 16 + iloc) * 32 + 16 + l15] = (_Float16)e1;
    }

    // re-layout P (C-frag -> A-frag) through wave-private LDS slice
    v16h pf;
#pragma unroll
    for (int t = 0; t < 8; ++t) {
      const int k = kmap(t, hi8);
      pf[2 * t]     = ps[(w * 16 + l15) * 32 + k];
      pf[2 * t + 1] = ps[(w * 16 + l15) * 32 + k + 1];
    }
    // V B-fragments (32 j-rows x 16 dims, two column tiles for d=32)
    v16h vf0, vf1;
#pragma unroll
    for (int t = 0; t < 8; ++t) {
      const int k = kmap(t, hi8);
      vf0[2 * t]     = vs[k * D_ + w * DH + l15];
      vf0[2 * t + 1] = vs[(k + 1) * D_ + w * DH + l15];
      vf1[2 * t]     = vs[k * D_ + w * DH + 16 + l15];
      vf1[2 * t + 1] = vs[(k + 1) * D_ + w * DH + 16 + l15];
    }
    o0 = __builtin_amdgcn_wmma_f32_16x16x32_f16(false, pf, false, vf0,
                                                (short)0, o0, false, false);
    o1 = __builtin_amdgcn_wmma_f32_16x16x32_f16(false, pf, false, vf1,
                                                (short)0, o1, false, false);
  }

  // normalize and write per-head output (concatenated heads layout [B,N,D])
#pragma unroll
  for (int r = 0; r < 8; ++r) {
    const int iloc = r + hi8;
    const float inv = 1.0f / lrow[r];
    float* op = ob + ((size_t)b * N_ + i0 + iloc) * D_ + w * DH;
    op[l15]      = o0[r] * inv;
    op[16 + l15] = o1[r] * inv;
  }
}

extern "C" void kernel_launch(void* const* d_in, const int* in_sizes, int n_in,
                              void* d_out, int out_size, void* d_ws, size_t ws_size,
                              hipStream_t stream) {
  const float* nodes = (const float*)d_in[0];
  const float* edges = (const float*)d_in[1];
  const int*   adj   = (const int*)d_in[2];
  const float* Wq    = (const float*)d_in[3];
  const float* Wk    = (const float*)d_in[4];
  const float* Wv    = (const float*)d_in[5];
  const float* We    = (const float*)d_in[6];
  const float* Wo    = (const float*)d_in[7];
  const float* bo    = (const float*)d_in[8];
  float* out = (float*)d_out;

  const size_t perBuf = (size_t)B_ * N_ * D_;  // 1,048,576 floats (4 MB)
  float* qb = (float*)d_ws;
  float* kb = qb + perBuf;
  float* vb = kb + perBuf;
  float* ob = vb + perBuf;   // total 16 MB of workspace

  const int M = B_ * N_;                       // 4096 rows, weights batch-shared
  const dim3 ggrid(M / 16, 2);
  gemm_nt_wmma<<<ggrid, 256, 0, stream>>>(nodes, Wq, nullptr, qb);
  gemm_nt_wmma<<<ggrid, 256, 0, stream>>>(nodes, Wk, nullptr, kb);
  gemm_nt_wmma<<<ggrid, 256, 0, stream>>>(nodes, Wv, nullptr, vb);

  const size_t smemBytes =
      (16 * 16 * 32 + 8 * 16 * 32) * sizeof(float) +   // es + bias
      16 * 32 * sizeof(int) +                          // adj tile
      (16 * D_ + 32 * D_ + 32 * D_ + 8 * 16 * 32) * sizeof(_Float16);  // qs/ks/vs/ps
  attn_fused<<<dim3(B_, N_ / 16), 256, smemBytes, stream>>>(qb, kb, vb, edges,
                                                            adj, We, ob);

  gemm_nt_wmma<<<ggrid, 256, 0, stream>>>(ob, Wo, bo, out);
}
